// TtFalconAttentionPrefill_16578573763575
// MI455X (gfx1250) — matrix operand order repaired and checked by reading, not compiled, not run on hardware
//
#include <hip/hip_runtime.h>
#include <hip/hip_bf16.h>

// ---------------------------------------------------------------------------
// Falcon attention prefill for MI455X (gfx1250, wave32, WMMA).
// Pipeline: f32->bf16 cvt, QKV GEMM (bf16 WMMA, f32 acc, double-buffered LDS,
// async global->LDS staging), RoPE, flash attention (online softmax, causal),
// dense GEMM -> f32 out.  Workspace usage: ~179 MB.
// ---------------------------------------------------------------------------

typedef __attribute__((ext_vector_type(16))) __bf16 v16bf;
typedef __attribute__((ext_vector_type(8)))  float  v8f;

union Frag {
  v16bf v;
  unsigned int u[8];
};

static constexpr int SEQ   = 2048;
static constexpr int NHEAD = 71;
static constexpr int HDIM  = 64;
static constexpr int HID   = NHEAD * HDIM;        // 4544
static constexpr int QKVN  = (NHEAD + 2) * HDIM;  // 4672

__device__ __forceinline__ unsigned short f2bf(float f) {
  unsigned int u = __float_as_uint(f);
  u += 0x7FFFu + ((u >> 16) & 1u);   // round-to-nearest-even
  return (unsigned short)(u >> 16);
}

// ---------------------------------------------------------------------------
// Async global->LDS copy of 32 contiguous bytes (2x B128), ASYNCcnt-tracked.
// Falls back to a synchronous VGPR round-trip if the builtin is unavailable.
// The builtin's pointer params are int-vector typed (per compiler diagnostic):
//   (global v4i*, lds v4i*, imm offset, imm cpol)
// ---------------------------------------------------------------------------
#if defined(__has_builtin)
#if __has_builtin(__builtin_amdgcn_global_load_async_to_lds_b128)
#define HAVE_ASYNC_LDS 1
#endif
#endif
#ifndef HAVE_ASYNC_LDS
#define HAVE_ASYNC_LDS 0
#endif

#if HAVE_ASYNC_LDS
typedef int vec4i __attribute__((vector_size(16)));
typedef __attribute__((address_space(1))) vec4i gvec4i;  // global
typedef __attribute__((address_space(3))) vec4i lvec4i;  // LDS
__device__ __forceinline__ void async_cp32(const void* g, void* l) {
  __builtin_amdgcn_global_load_async_to_lds_b128((gvec4i*)g, (lvec4i*)l, 0, 0);
  __builtin_amdgcn_global_load_async_to_lds_b128((gvec4i*)g, (lvec4i*)l, 16, 0);
}
__device__ __forceinline__ void async_wait() {
  asm volatile("s_wait_asynccnt 0x0" ::: "memory");
}
#else
__device__ __forceinline__ void async_cp32(const void* g, void* l) {
  uint4 a = ((const uint4*)g)[0];
  uint4 b = ((const uint4*)g)[1];
  ((uint4*)l)[0] = a;
  ((uint4*)l)[1] = b;
}
__device__ __forceinline__ void async_wait() {}
#endif

// ---------------------------------------------------------------------------
// f32 -> bf16 elementwise (vectorized x4)
// ---------------------------------------------------------------------------
__global__ __launch_bounds__(256) void cvt_f32_to_bf16(
    const float* __restrict__ src, unsigned short* __restrict__ dst, int n4) {
  int i = blockIdx.x * blockDim.x + threadIdx.x;
  if (i >= n4) return;
  float4 f = ((const float4*)src)[i];
  uint2 p;
  p.x = (unsigned int)f2bf(f.x) | ((unsigned int)f2bf(f.y) << 16);
  p.y = (unsigned int)f2bf(f.z) | ((unsigned int)f2bf(f.w) << 16);
  ((uint2*)dst)[i] = p;
}

// ---------------------------------------------------------------------------
// GEMM: C[f32, MxN] = A[bf16, MxK row-major] * B[bf16, KxN row-major]
// Block tile 128x64, BK=32, double-buffered LDS, 1 barrier / K-step.
// 256 threads = 8 waves in 4x2; each wave computes a 32x32 tile (4 WMMA acc).
// ---------------------------------------------------------------------------
static constexpr int LDA = 40;  // 32 + 8 halfs pad (bank spread)
static constexpr int LDB = 40;

__global__ __launch_bounds__(256) void gemm_bf16_f32(
    const unsigned short* __restrict__ A, const unsigned short* __restrict__ B,
    float* __restrict__ C, int M, int N, int K) {
  (void)M;
  __shared__ unsigned short As[2][128][LDA];  // [buf][m][k]
  __shared__ unsigned short Bs[2][64][LDB];   // [buf][n][k] (transposed)

  const int tid  = threadIdx.x;
  const int lane = tid & 31;
  const int w    = tid >> 5;
  const int wm   = w >> 1;          // 0..3
  const int wn   = w & 1;           // 0..1
  const int ln   = lane & 15;
  const int hi   = lane >> 4;       // 0 or 1
  const int koff = hi ? 8 : 0;      // A-frag K base per half-wave
  const int mBase = blockIdx.y * 128;
  const int nBase = blockIdx.x * 64;

  v8f zero = {};
  v8f acc[2][2];
  acc[0][0] = zero; acc[0][1] = zero; acc[1][0] = zero; acc[1][1] = zero;

  const int arow = tid >> 1, aseg = tid & 1;   // A stage: 2 thr/row, 16 halfs
  const int brow = tid >> 3, bseg = tid & 7;   // B stage: 8 thr/row, 8 halfs
  const unsigned short* Abase = A + (size_t)(mBase + arow) * K + aseg * 16;
  const unsigned short* Bbase = B + (size_t)brow * N + nBase + bseg * 8;

  // ---- prologue: stage K-step 0 into buffer 0 ----
  async_cp32(Abase, &As[0][arow][aseg * 16]);
  {
    uint4 bv = *(const uint4*)Bbase;
    const unsigned short* bh = (const unsigned short*)&bv;
#pragma unroll
    for (int e = 0; e < 8; ++e) Bs[0][bseg * 8 + e][brow] = bh[e];
  }
  async_wait();
  __syncthreads();

  int buf = 0;
  for (int k0 = 0; k0 < K; k0 += 32) {
    const bool next = (k0 + 32) < K;   // block-uniform
    uint4 bv = {};
    if (next) {
      // kick off next tile: A async -> LDS, B global -> VGPR (early)
      async_cp32(Abase + k0 + 32, &As[buf ^ 1][arow][aseg * 16]);
      bv = *(const uint4*)(Bbase + (size_t)(k0 + 32) * N);
    }

    // ---- build fragments from current buffer ----
    Frag a0, a1, b0, b1;
    const unsigned short* r0 = &As[buf][wm * 32 + 0  + ln][0];
    const unsigned short* r1 = &As[buf][wm * 32 + 16 + ln][0];
#pragma unroll
    for (int j = 0; j < 4; ++j) {
      a0.u[j]     = *(const unsigned int*)(r0 + koff + 2 * j);
      a0.u[j + 4] = *(const unsigned int*)(r0 + 16 + koff + 2 * j);
      a1.u[j]     = *(const unsigned int*)(r1 + koff + 2 * j);
      a1.u[j + 4] = *(const unsigned int*)(r1 + 16 + koff + 2 * j);
    }
    const unsigned short* c0 = &Bs[buf][wn * 32 + 0  + ln][hi * 8];
    const unsigned short* c1 = &Bs[buf][wn * 32 + 16 + ln][hi * 8];
#pragma unroll
    for (int j = 0; j < 8; ++j) {
      b0.u[j] = *(const unsigned int*)(c0 + 2 * j + ((j & 4) ? 8 : 0));
      b1.u[j] = *(const unsigned int*)(c1 + 2 * j + ((j & 4) ? 8 : 0));
    }

    acc[0][0] = __builtin_amdgcn_wmma_f32_16x16x32_bf16(false, a0.v, false, b0.v, (short)0, acc[0][0], false, false);
    acc[0][1] = __builtin_amdgcn_wmma_f32_16x16x32_bf16(false, a0.v, false, b1.v, (short)0, acc[0][1], false, false);
    acc[1][0] = __builtin_amdgcn_wmma_f32_16x16x32_bf16(false, a1.v, false, b0.v, (short)0, acc[1][0], false, false);
    acc[1][1] = __builtin_amdgcn_wmma_f32_16x16x32_bf16(false, a1.v, false, b1.v, (short)0, acc[1][1], false, false);

    if (next) {
      // scatter-transpose B into the other buffer (no conflict with readers)
      const unsigned short* bh = (const unsigned short*)&bv;
#pragma unroll
      for (int e = 0; e < 8; ++e) Bs[buf ^ 1][bseg * 8 + e][brow] = bh[e];
    }
    async_wait();
    __syncthreads();
    buf ^= 1;
  }

  // ---- epilogue: C layout VGPR r -> (M=r | r+8 by half-wave, N=ln) ----
#pragma unroll
  for (int i = 0; i < 2; ++i)
#pragma unroll
    for (int j = 0; j < 2; ++j)
#pragma unroll
      for (int r = 0; r < 8; ++r) {
        int row = mBase + wm * 32 + i * 16 + r + hi * 8;
        int col = nBase + wn * 32 + j * 16 + ln;
        C[(size_t)row * N + col] = acc[i][j][r];
      }
}

// ---------------------------------------------------------------------------
// RoPE + split/transposed repack:
//   qT[h][s][d] (bf16, roped), k[s][d] (bf16, roped), v[s][d] (bf16)
// ---------------------------------------------------------------------------
__global__ __launch_bounds__(256) void rope_split(
    const float* __restrict__ qkv, unsigned short* __restrict__ qT,
    unsigned short* __restrict__ kmat, unsigned short* __restrict__ vmat) {
  const int s   = blockIdx.x;
  const int tid = threadIdx.x;
  const float* row = qkv + (size_t)s * QKVN;
  const int d = tid & 63;
  const int j = d & 31;
  float sn, cs;
  float inv = __expf(-(float)j * 0.2878231366242557f);  // ln(10000)/32
  __sincosf((float)s * inv, &sn, &cs);
  const float sgn = (d < 32) ? -1.0f : 1.0f;
  const int   prt = (d < 32) ? d + 32 : d - 32;

  for (int h = tid >> 6; h < NHEAD; h += 4) {
    float x  = row[h * HDIM + d];
    float xp = row[h * HDIM + prt];
    qT[((size_t)h * SEQ + s) * HDIM + d] = f2bf(x * cs + sgn * xp * sn);
  }
  if (tid < 64) {
    float x  = row[HID + d];
    float xp = row[HID + prt];
    kmat[(size_t)s * HDIM + d] = f2bf(x * cs + sgn * xp * sn);
  } else if (tid < 128) {
    vmat[(size_t)s * HDIM + d] = f2bf(row[HID + HDIM + d]);
  }
}

// ---------------------------------------------------------------------------
// Flash attention, causal, multi-query (one shared K/V head).
// Grid: (SEQ/128, NHEAD). 8 waves; each wave owns 16 query rows.
// ---------------------------------------------------------------------------
__global__ __launch_bounds__(256) void attn_flash(
    const unsigned short* __restrict__ qT, const unsigned short* __restrict__ kmat,
    const unsigned short* __restrict__ vmat, unsigned short* __restrict__ attn) {
  __shared__ unsigned short Ks[64][72];       // [t][d]   (B-frags for scores)
  __shared__ unsigned short Vt[64][72];       // [d][t]   (B-frags for P@V)
  __shared__ unsigned short Ps[8][16][72];    // per-wave probs (C->A relayout)

  const int tid  = threadIdx.x;
  const int lane = tid & 31;
  const int w    = tid >> 5;
  const int ln   = lane & 15;
  const int hi   = lane >> 4;
  const int koff = hi ? 8 : 0;
  const int qb   = blockIdx.x;
  const int h    = blockIdx.y;
  const int qbase = qb * 128 + w * 16;

  // q A-fragments for this wave's 16 rows, both K-steps of HDIM=64
  Frag qA[2];
  {
    const unsigned short* qp = qT + ((size_t)h * SEQ + qbase + ln) * HDIM;
#pragma unroll
    for (int ks = 0; ks < 2; ++ks)
#pragma unroll
      for (int jj = 0; jj < 4; ++jj) {
        qA[ks].u[jj]     = *(const unsigned int*)(qp + ks * 32 + koff + 2 * jj);
        qA[ks].u[jj + 4] = *(const unsigned int*)(qp + ks * 32 + 16 + koff + 2 * jj);
      }
  }

  v8f zero = {};
  v8f accO[4];
  accO[0] = zero; accO[1] = zero; accO[2] = zero; accO[3] = zero;
  float m[8], l[8];
#pragma unroll
  for (int r = 0; r < 8; ++r) { m[r] = -3.0e38f; l[r] = 0.f; }

  const int kbmax = qb * 2 + 1;  // key blocks up to the causal diagonal
  for (int kb = 0; kb <= kbmax; ++kb) {
    // ---- stage K tile row-major (async global->LDS) ----
    {
      int row = tid >> 2, seg = tid & 3;
      async_cp32(kmat + (size_t)(kb * 64 + row) * HDIM + seg * 16,
                 &Ks[row][seg * 16]);
    }
    // ---- stage V tile transposed ----
    {
      int trow = tid & 63, dseg = tid >> 6;
      const unsigned short* g = vmat + (size_t)(kb * 64 + trow) * HDIM + dseg * 16;
#pragma unroll
      for (int e = 0; e < 16; ++e) Vt[dseg * 16 + e][trow] = g[e];
    }
    async_wait();
    __syncthreads();

    if (kb * 64 <= qbase + 15) {  // wave-uniform: EXEC all-ones inside
      // ---- scores: 16 rows x 64 keys, f32 acc ----
      v8f st[4];
#pragma unroll
      for (int ct = 0; ct < 4; ++ct) {
        v8f sc = zero;
#pragma unroll
        for (int ks = 0; ks < 2; ++ks) {
          Frag b;
          const unsigned short* cp = &Ks[ct * 16 + ln][ks * 32 + hi * 8];
#pragma unroll
          for (int jj = 0; jj < 8; ++jj) b.u[jj] = *(const unsigned int*)(cp + 2 * jj + ((jj & 4) ? 8 : 0));
          sc = __builtin_amdgcn_wmma_f32_16x16x32_bf16(false, qA[ks].v, false, b.v, (short)0, sc, false, false);
        }
        st[ct] = sc;
      }
      // ---- scale + causal mask ----
#pragma unroll
      for (int ct = 0; ct < 4; ++ct)
#pragma unroll
        for (int r = 0; r < 8; ++r) {
          int qrow = qbase + r + hi * 8;
          int kcol = kb * 64 + ct * 16 + ln;
          float sv = st[ct][r] * 0.125f;   // 1/sqrt(64)
          st[ct][r] = (kcol > qrow) ? -1.0e9f : sv;
        }
      // ---- online softmax (per-row stats land in the right lanes) ----
#pragma unroll
      for (int r = 0; r < 8; ++r) {
        float rmax = fmaxf(fmaxf(st[0][r], st[1][r]), fmaxf(st[2][r], st[3][r]));
#pragma unroll
        for (int off = 1; off < 16; off <<= 1)
          rmax = fmaxf(rmax, __shfl_xor(rmax, off, 32));
        float mnew = fmaxf(m[r], rmax);
        float corr = __expf(m[r] - mnew);
        float rsum = 0.f;
#pragma unroll
        for (int ct = 0; ct < 4; ++ct) {
          float p = __expf(st[ct][r] - mnew);
          rsum += p;
          Ps[w][r + hi * 8][ct * 16 + ln] = f2bf(p);
        }
#pragma unroll
        for (int off = 1; off < 16; off <<= 1)
          rsum += __shfl_xor(rsum, off, 32);
        l[r] = l[r] * corr + rsum;
        m[r] = mnew;
#pragma unroll
        for (int dt = 0; dt < 4; ++dt) accO[dt][r] *= corr;
      }
      // ---- accO += P @ V ----
#pragma unroll
      for (int ks = 0; ks < 2; ++ks) {
        Frag pA;
        const unsigned short* rp = &Ps[w][ln][ks * 32];
#pragma unroll
        for (int jj = 0; jj < 4; ++jj) {
          pA.u[jj]     = *(const unsigned int*)(rp + koff + 2 * jj);
          pA.u[jj + 4] = *(const unsigned int*)(rp + 16 + koff + 2 * jj);
        }
#pragma unroll
        for (int dt = 0; dt < 4; ++dt) {
          Frag b;
          const unsigned short* cp = &Vt[dt * 16 + ln][ks * 32 + hi * 8];
#pragma unroll
          for (int jj = 0; jj < 8; ++jj) b.u[jj] = *(const unsigned int*)(cp + 2 * jj + ((jj & 4) ? 8 : 0));
          accO[dt] = __builtin_amdgcn_wmma_f32_16x16x32_bf16(false, pA.v, false, b.v, (short)0, accO[dt], false, false);
        }
      }
    }
    __syncthreads();
  }

  // ---- normalize + store bf16 attn [SEQ][HID] (A-ready for dense GEMM) ----
#pragma unroll
  for (int dt = 0; dt < 4; ++dt)
#pragma unroll
    for (int r = 0; r < 8; ++r) {
      float o = accO[dt][r] / l[r];
      int row = qbase + r + hi * 8;
      int col = h * HDIM + dt * 16 + ln;
      attn[(size_t)row * HID + col] = f2bf(o);
    }
}

// ---------------------------------------------------------------------------
extern "C" void kernel_launch(void* const* d_in, const int* in_sizes, int n_in,
                              void* d_out, int out_size, void* d_ws, size_t ws_size,
                              hipStream_t stream) {
  (void)in_sizes; (void)n_in; (void)out_size; (void)ws_size;
  const float* x  = (const float*)d_in[0];
  // d_in[1] = attention_mask: causal, applied analytically in attn_flash
  const float* qw = (const float*)d_in[2];
  const float* dw = (const float*)d_in[3];
  float* out = (float*)d_out;

  char* ws = (char*)d_ws;
  size_t off = 0;
  auto alloc = [&](size_t bytes) {
    char* p = ws + off;
    off += (bytes + 255) & ~(size_t)255;
    return p;
  };
  unsigned short* xb  = (unsigned short*)alloc((size_t)SEQ * HID * 2);    // 18.6 MB
  unsigned short* w1b = (unsigned short*)alloc((size_t)HID * QKVN * 2);   // 42.5 MB
  unsigned short* w2b = (unsigned short*)alloc((size_t)HID * HID * 2);    // 41.3 MB
  float*          qkv = (float*)alloc((size_t)SEQ * QKVN * 4);            // 38.3 MB
  unsigned short* qT  = (unsigned short*)alloc((size_t)NHEAD * SEQ * HDIM * 2); // 18.6 MB
  unsigned short* kb  = (unsigned short*)alloc((size_t)SEQ * HDIM * 2);   // 0.26 MB
  unsigned short* vb  = (unsigned short*)alloc((size_t)SEQ * HDIM * 2);   // 0.26 MB
  unsigned short* ab  = (unsigned short*)alloc((size_t)SEQ * HID * 2);    // 18.6 MB
  // total ~179 MB

  const int n1 = SEQ * HID / 4, n2 = HID * QKVN / 4, n3 = HID * HID / 4;
  cvt_f32_to_bf16<<<(n1 + 255) / 256, 256, 0, stream>>>(x, xb, n1);
  cvt_f32_to_bf16<<<(n2 + 255) / 256, 256, 0, stream>>>(qw, w1b, n2);
  cvt_f32_to_bf16<<<(n3 + 255) / 256, 256, 0, stream>>>(dw, w2b, n3);

  gemm_bf16_f32<<<dim3(QKVN / 64, SEQ / 128), 256, 0, stream>>>(xb, w1b, qkv, SEQ, QKVN, HID);
  rope_split<<<SEQ, 256, 0, stream>>>(qkv, qT, kb, vb);
  attn_flash<<<dim3(SEQ / 128, NHEAD), 256, 0, stream>>>(qT, kb, vb, ab);
  gemm_bf16_f32<<<dim3(HID / 64, SEQ / 128), 256, 0, stream>>>(ab, w2b, out, SEQ, HID, HID);
}
